// SpectralAttention_73212012527750
// MI455X (gfx1250) — matrix operand order repaired and checked by reading, not compiled, not run on hardware
//
#include <hip/hip_runtime.h>
#include <cstdint>
#include <cstddef>

// ---------------------------------------------------------------------------
// SpectralAttention for MI455X (gfx1250, wave32, WMMA 16x16x32 f16).
//   xt   = sum_k Toeplitz(basis_k) @ (x @ Wm_k) + bm
//   g_t  = lrelu2( v_t^T (kvs .* Wgz) k_t + bgz ) + eps
//   ctxt = c * [tril(Q K^T) * g_col] @ V / (cumsum(g) + eps)   (c = kvs[0])
//   out  = (ctxt * sigmoid(sim*Wg+bg)) @ Wo + bo
// GEMMs stage B panels into LDS with global_load_async_to_lds_b128 (ASYNCcnt);
// Toeplitz A-tiles are pre-baked into WMMA fragment layout (Atab).
// ---------------------------------------------------------------------------

typedef _Float16 half_t;
typedef __attribute__((ext_vector_type(16))) _Float16 v16h;
typedef __attribute__((ext_vector_type(8)))  _Float16 v8h;
typedef __attribute__((ext_vector_type(8)))  float    v8f;

constexpr int B_ = 2, L_ = 1024, D_ = 512, H_ = 8, K_ = 24, h_ = 64;
constexpr int LP_ = L_ + 32;           // padded basis row (32 leading zeros)
constexpr int ND16_ = 64;              // distinct (l0-s0)/16 values
constexpr float EPS_ = 1e-5f;

#define DEVFN static __device__ __forceinline__

DEVFN v8f wmma16(v16h a, v16h b, v8f c) {
  return __builtin_amdgcn_wmma_f32_16x16x32_f16(false, a, false, b, (short)0, c,
                                                false, false);
}

DEVFN v16h cat8(v8h lo, v8h hi) {
  return __builtin_shufflevector(lo, hi, 0, 1, 2, 3, 4, 5, 6, 7, 8, 9, 10, 11,
                                 12, 13, 14, 15);
}

// Load 16 contiguous halves (32B, 16B-aligned) as one fragment.
DEVFN v16h load_contig16(const half_t* p) {
  return cat8(*(const v8h*)p, *(const v8h*)(p + 8));
}

// A fragment: 16x32 f16 tile, row-major source (16B-aligned rows).
// lanes 0-15: M=lane, K=0..15; lanes 16-31: K=16..31
DEVFN v16h load_a_frag(const half_t* A, int lda, int lane) {
  const int hs = lane >> 4, m = lane & 15;
  const half_t* base = A + (size_t)m * lda + hs * 16;
  return cat8(*(const v8h*)base, *(const v8h*)(base + 8));
}

// ---------------------------------------------------------------------------
// WMMA GEMM: 8 waves/block share one 16-col B panel staged in LDS via async
// copy. C written as f32 and/or f16 row-major and/or f16 transposed.
// Assumes M%128==0, N%16==0, Kd%32==0, Kd<=512.
// ---------------------------------------------------------------------------
__global__ void __launch_bounds__(256)
wmma_gemm(const half_t* __restrict__ A, int lda, size_t aStrideZ,
          const half_t* __restrict__ B, int ldb, size_t bStrideZ,
          float* __restrict__ Cf, half_t* __restrict__ Ch,
          half_t* __restrict__ ChT, int ldc, int ldct, size_t cStrideZ,
          const float* __restrict__ bias, int M, int N, int Kd) {
  __shared__ half_t ldsRaw[512 * 16];   // B panel, row-major (K x 16)
  __shared__ half_t ldsT[16 * 512];     // B panel, transposed (16 x K)
  A += (size_t)blockIdx.y * aStrideZ;
  B += (size_t)blockIdx.y * bStrideZ;
  const int tid = threadIdx.x, lane = tid & 31, wslot = tid >> 5;
  const int ntN = N >> 4;
  const int tn = blockIdx.x % ntN;
  const int tm = (blockIdx.x / ntN) * 8 + wslot;

  // --- stage B panel (Kd x 16 halves) into LDS with async b128 copies ---
  const half_t* Bcol = B + tn * 16;
  const int nxfer = (Kd * 2) >> 8;      // (Kd rows * 2 chunks) / 256 threads
  for (int it = 0; it < nxfer; ++it) {
    const int id = it * 256 + tid;
    const int kk = id >> 1, pp = id & 1;
    const half_t* gsrc = Bcol + (size_t)kk * ldb + pp * 8;
    const uint32_t ldst = (uint32_t)(uintptr_t)(&ldsRaw[kk * 16 + pp * 8]);
    asm volatile("global_load_async_to_lds_b128 %0, %1, off"
                 :: "v"(ldst), "v"(gsrc)
                 : "memory");
  }
  asm volatile("s_wait_asynccnt 0x0" ::: "memory");
  __syncthreads();
  // --- transpose panel inside LDS so B-frags are contiguous-K ---
  const int nel = (Kd * 16) >> 8;
  for (int it = 0; it < nel; ++it) {
    const int id = it * 256 + tid;
    const int n = id & 15, kk = id >> 4;
    ldsT[n * Kd + kk] = ldsRaw[kk * 16 + n];
  }
  __syncthreads();

  const half_t* Abase = A + (size_t)(tm * 16) * lda;
  const half_t* bt = &ldsT[(size_t)(lane & 15) * Kd + ((lane >> 4) * 16)];
  v8f acc = {};
  for (int k0 = 0; k0 < Kd; k0 += 32) {
    __builtin_prefetch(Abase + (size_t)(lane & 15) * lda + k0 + 64, 0, 1);
    v16h a = load_a_frag(Abase + k0, lda, lane);
    v16h b = load_contig16(bt + k0);
    acc = wmma16(a, b, acc);
  }
  const int n = tn * 16 + (lane & 15);
  const float bv = bias ? bias[n] : 0.f;
  const int hs = lane >> 4;
  const size_t zo = (size_t)blockIdx.y * cStrideZ;
#pragma unroll
  for (int r = 0; r < 8; ++r) {
    const int m = tm * 16 + hs * 8 + r;
    const float v = acc[r] + bv;
    if (Cf) Cf[zo + (size_t)m * ldc + n] = v;
    if (Ch) Ch[zo + (size_t)m * ldc + n] = (half_t)v;
    if (ChT) ChT[zo + (size_t)n * ldct + m] = (half_t)v;
  }
}

// ---------------------------------------------------------------------------
// Pre-bake Toeplitz A-tiles in WMMA fragment layout:
//   Atab[((k*64 + d16)*32 + lane)*16 + e] = basisPad[k][d16*16 + (lane&15)
//                                            - ((lane>>4)*16 + e)]
// ---------------------------------------------------------------------------
__global__ void build_atab(const half_t* __restrict__ basisT,
                           half_t* __restrict__ Atab) {
  const int i = blockIdx.x * blockDim.x + threadIdx.x;
  if (i >= K_ * ND16_ * 32 * 16) return;
  const int e = i & 15;
  const int lane = (i >> 4) & 31;
  const int d16 = (i >> 9) & (ND16_ - 1);
  const int k = i >> 15;
  const int hs = lane >> 4, mr = lane & 15;
  const int kk = hs * 16 + e;
  const int dlt = d16 * 16 + mr - kk;          // >= -31, padded
  Atab[i] = basisT[k * LP_ + 32 + dlt];
}

// ---------------------------------------------------------------------------
// Causal Toeplitz combine: xt[b,l,d] = sum_k sum_{s<=l} basis[l-s,k]*P_k[b,s,d]
// P stored transposed per k: Pt[k][d][b*L+s]. A-frags from Atab (contiguous).
// ---------------------------------------------------------------------------
__global__ void __launch_bounds__(256)
combine_kernel(const half_t* __restrict__ Pt, const half_t* __restrict__ Atab,
               const float* __restrict__ bm, half_t* __restrict__ xt_h) {
  const int lane = threadIdx.x & 31;
  const int wave = (blockIdx.x * blockDim.x + threadIdx.x) >> 5;
  const int ntD = D_ / 16, ntL = L_ / 16;
  if (wave >= B_ * ntL * ntD) return;
  const int b = wave / (ntL * ntD);
  const int rem = wave % (ntL * ntD);
  const int l0 = (rem / ntD) * 16, d0 = (rem % ntD) * 16;
  const int hs = lane >> 4, mr = lane & 15;
  const size_t BLD = (size_t)B_ * L_ * D_;
  v8f acc = {};
  for (int k = 0; k < K_; ++k) {
    const half_t* atk = Atab + ((size_t)k * ND16_ * 32 + lane) * 16;
    const half_t* Pk =
        Pt + (size_t)k * BLD + (size_t)(d0 + mr) * (B_ * L_) + (size_t)b * L_;
    for (int s0 = 0; s0 <= l0; s0 += 32) {
      const int d16 = (l0 - s0) >> 4;
      v16h a = load_contig16(atk + (size_t)d16 * (32 * 16));
      v16h bf = load_contig16(Pk + s0 + hs * 16);
      acc = wmma16(a, bf, acc);
    }
  }
  const int n = d0 + mr;
  const float bv = bm[n];
#pragma unroll
  for (int r = 0; r < 8; ++r) {
    const int m = l0 + hs * 8 + r;
    xt_h[((size_t)b * L_ + m) * D_ + n] = (half_t)(acc[r] + bv);
  }
}

// ---------------------------------------------------------------------------
// Elementwise helpers
// ---------------------------------------------------------------------------
__global__ void cvt_f32_to_f16(const float* __restrict__ in,
                               half_t* __restrict__ out, int n) {
  const int i = blockIdx.x * blockDim.x + threadIdx.x;
  if (i < n) out[i] = (half_t)in[i];
}

__global__ void basis_transpose(const float* __restrict__ basis,
                                half_t* __restrict__ basisT) {
  const int i = blockIdx.x * blockDim.x + threadIdx.x;
  if (i < K_ * LP_) {
    const int k = i / LP_, lp = i % LP_;
    basisT[i] = (lp < 32) ? (half_t)0.f : (half_t)basis[(lp - 32) * K_ + k];
  }
}

// MgzT[j*h + i] = kvs[i*h+j] * Wgz[i*h+j]  (transposed for contiguous reads)
__global__ void mgz_kernel(const float* __restrict__ kvs,
                           const float* __restrict__ Wgz,
                           float* __restrict__ MgzT) {
  const int idx = blockIdx.x * blockDim.x + threadIdx.x;
  if (idx < h_ * h_) {
    const int j = idx >> 6, i = idx & 63;
    MgzT[idx] = kvs[i * h_ + j] * Wgz[i * h_ + j];
  }
}

// g[b,h,l] = lrelu2(v^T (kvs.*Wgz) k + bgz) + eps ; sim = (q.k)*qk_scale[h]
__global__ void __launch_bounds__(256)
gate_kernel(const half_t* __restrict__ q, const half_t* __restrict__ k,
            const half_t* __restrict__ v, const float* __restrict__ MgzT,
            const float* __restrict__ bgz, const float* __restrict__ qks,
            float* __restrict__ g, float* __restrict__ sim) {
  const int wave = (blockIdx.x * blockDim.x + threadIdx.x) >> 5;
  const int lane = threadIdx.x & 31;
  if (wave >= B_ * H_ * L_) return;
  const int l = wave & (L_ - 1);
  const int bh = wave >> 10;           // L_ == 1024
  const int head = bh & (H_ - 1);
  const int b = bh >> 3;               // H_ == 8
  const half_t* qp = q + ((size_t)(b * L_ + l)) * D_ + head * h_;
  const half_t* kp = k + ((size_t)(b * L_ + l)) * D_ + head * h_;
  const half_t* vp = v + ((size_t)(b * L_ + l)) * D_ + head * h_;
  float ag = 0.f, as = 0.f;
#pragma unroll
  for (int jj = 0; jj < 2; ++jj) {
    const int j = lane + jj * 32;
    const float kj = (float)kp[j], qj = (float)qp[j];
    as += qj * kj;
    const float* mrow = MgzT + j * h_;
    float s = 0.f;
    for (int i = 0; i < h_; ++i) s += (float)vp[i] * mrow[i];
    ag += s * kj;
  }
#pragma unroll
  for (int off = 16; off > 0; off >>= 1) {
    ag += __shfl_xor(ag, off, 32);
    as += __shfl_xor(as, off, 32);
  }
  if (lane == 0) {
    const float gl = ag + bgz[0];
    const float lr = gl > 0.f ? gl : 0.01f * gl;
    g[wave] = lr * lr + EPS_;
    sim[wave] = as * qks[head];
  }
}

// inclusive scan of g over L per (b,h)
__global__ void __launch_bounds__(1024)
cumsum_kernel(const float* __restrict__ g, float* __restrict__ cumg) {
  __shared__ float buf[L_];
  const int bh = blockIdx.x, t = threadIdx.x;
  buf[t] = g[(size_t)bh * L_ + t];
  __syncthreads();
  for (int off = 1; off < L_; off <<= 1) {
    const float add = (t >= off) ? buf[t - off] : 0.f;
    __syncthreads();
    buf[t] += add;
    __syncthreads();
  }
  cumg[(size_t)bh * L_ + t] = buf[t];
}

// ---------------------------------------------------------------------------
// Causal gated linear attention. One wave per (b, head, 16-row l-tile).
// vT is the transposed V copy: vT[d*2048 + b*1024 + t].
// ---------------------------------------------------------------------------
__global__ void __launch_bounds__(256)
attn_kernel(const half_t* __restrict__ q, const half_t* __restrict__ k,
            const half_t* __restrict__ vT, const float* __restrict__ g,
            const float* __restrict__ cumg, const float* __restrict__ sim,
            const float* __restrict__ kvs, const float* __restrict__ Wg,
            const float* __restrict__ bg, half_t* __restrict__ outpre) {
  __shared__ half_t sbuf[8][16 * 32];
  const int wslot = threadIdx.x >> 5;
  const int lane = threadIdx.x & 31;
  const int wave = blockIdx.x * 8 + wslot;
  const int ntL = L_ / 16;
  if (wave >= B_ * H_ * ntL) return;
  const int tl = wave % ntL;
  const int bh = wave / ntL;
  const int head = bh % H_;
  const int b = bh / H_;
  const int l0 = tl * 16;
  const half_t* qb = q + (size_t)b * L_ * D_ + head * h_;
  const half_t* kb = k + (size_t)b * L_ * D_ + head * h_;
  const float* gb = g + (size_t)bh * L_;
  const int hs = lane >> 4, mr = lane & 15;
  v8f o0 = {}, o1 = {}, o2 = {}, o3 = {};
  for (int t0 = 0; t0 <= l0; t0 += 32) {
    // --- S tile (16 l-rows x 32 t-cols) ---
#pragma unroll
    for (int sub = 0; sub < 2; ++sub) {
      const int tc0 = t0 + sub * 16;
      v8f s = {};
#pragma unroll
      for (int c0 = 0; c0 < h_; c0 += 32) {
        v16h a = load_a_frag(qb + (size_t)l0 * D_ + c0, D_, lane);
        // B[kk,n] = K[tc0+n, c0+kk] : contiguous 16 halves per lane
        v16h bf = load_contig16(kb + (size_t)(tc0 + mr) * D_ + c0 + hs * 16);
        s = wmma16(a, bf, s);
      }
      const float gt = gb[tc0 + mr];
      const int t = tc0 + mr;
#pragma unroll
      for (int r = 0; r < 8; ++r) {
        const int m = hs * 8 + r;
        const float val = (t <= l0 + m) ? s[r] * gt : 0.f;
        sbuf[wslot][m * 32 + sub * 16 + mr] = (half_t)val;
      }
    }
    // --- O += S @ V ---  (same-wave LDS RAW: DS ops are in-order)
    v16h a2 = load_a_frag(&sbuf[wslot][0], 32, lane);
#pragma unroll
    for (int nc = 0; nc < 4; ++nc) {
      const half_t* vr = vT + (size_t)(head * h_ + nc * 16 + mr) * (B_ * L_) +
                         (size_t)b * L_ + t0 + hs * 16;
      v16h bf = load_contig16(vr);
      v8f& o = nc == 0 ? o0 : nc == 1 ? o1 : nc == 2 ? o2 : o3;
      o = wmma16(a2, bf, o);
    }
  }
  // --- epilogue: normalize, sigmoid gate, store fp16 (B,L,D) ---
  const float c = kvs[0];
  const float* cgb = cumg + (size_t)bh * L_;
  const float* simb = sim + (size_t)bh * L_;
#pragma unroll
  for (int nc = 0; nc < 4; ++nc) {
    const v8f& o = nc == 0 ? o0 : nc == 1 ? o1 : nc == 2 ? o2 : o3;
    const int i = nc * 16 + mr;
    const float wgi = Wg[i], bgi = bg[i];
#pragma unroll
    for (int r = 0; r < 8; ++r) {
      const int l = l0 + hs * 8 + r;
      const float ctxt = c * o[r] / (cgb[l] + EPS_);
      const float go = simb[l] * wgi + bgi;
      const float val = ctxt * (1.f / (1.f + expf(-go)));
      outpre[((size_t)b * L_ + l) * D_ + head * h_ + i] = (half_t)val;
    }
  }
}

// ---------------------------------------------------------------------------
// Host-side orchestration
// ---------------------------------------------------------------------------
extern "C" void kernel_launch(void* const* d_in, const int* in_sizes, int n_in,
                              void* d_out, int out_size, void* d_ws,
                              size_t ws_size, hipStream_t stream) {
  const float* x    = (const float*)d_in[0];
  const float* basis= (const float*)d_in[1];
  const float* Wq   = (const float*)d_in[2];
  const float* bq   = (const float*)d_in[3];
  const float* Wk   = (const float*)d_in[4];
  const float* bk   = (const float*)d_in[5];
  const float* Wv   = (const float*)d_in[6];
  const float* bv   = (const float*)d_in[7];
  const float* Wo   = (const float*)d_in[8];
  const float* bo   = (const float*)d_in[9];
  const float* Wm   = (const float*)d_in[10];
  const float* bm   = (const float*)d_in[11];
  const float* Wgz  = (const float*)d_in[12];
  const float* bgz  = (const float*)d_in[13];
  const float* Wg   = (const float*)d_in[14];
  const float* bg   = (const float*)d_in[15];
  const float* kvs  = (const float*)d_in[16];
  const float* qks  = (const float*)d_in[17];
  float* out = (float*)d_out;

  char* p = (char*)d_ws;
  auto alloc_h = [&](size_t n) {
    half_t* r = (half_t*)p; p += ((n * 2 + 255) & ~(size_t)255); return r;
  };
  auto alloc_f = [&](size_t n) {
    float* r = (float*)p; p += ((n * 4 + 255) & ~(size_t)255); return r;
  };

  const size_t BLD = (size_t)B_ * L_ * D_;          // 1,048,576
  half_t* x_h    = alloc_h(BLD);
  half_t* Wm_h   = alloc_h((size_t)K_ * D_ * D_);
  half_t* Wq_h   = alloc_h((size_t)D_ * D_);
  half_t* Wk_h   = alloc_h((size_t)D_ * D_);
  half_t* Wv_h   = alloc_h((size_t)D_ * D_);
  half_t* Wo_h   = alloc_h((size_t)D_ * D_);
  half_t* basisT = alloc_h((size_t)K_ * LP_);
  half_t* Atab   = alloc_h((size_t)K_ * ND16_ * 32 * 16);
  half_t* P_h    = alloc_h((size_t)K_ * BLD);       // transposed per k
  half_t* xt_h   = alloc_h(BLD);
  half_t* q_h    = alloc_h(BLD);
  half_t* k_h    = alloc_h(BLD);
  half_t* v_h    = alloc_h(BLD);
  half_t* vT_h   = alloc_h(BLD);
  half_t* op_h   = alloc_h(BLD);
  float*  MgzT   = alloc_f((size_t)h_ * h_);
  float*  gbuf   = alloc_f((size_t)B_ * H_ * L_);
  float*  cgbuf  = alloc_f((size_t)B_ * H_ * L_);
  float*  simbuf = alloc_f((size_t)B_ * H_ * L_);
  if ((size_t)(p - (char*)d_ws) > ws_size) return;  // insufficient scratch

  auto cvt = [&](const float* src, half_t* dst, int n) {
    cvt_f32_to_f16<<<(n + 255) / 256, 256, 0, stream>>>(src, dst, n);
  };
  cvt(x, x_h, (int)BLD);
  cvt(Wm, Wm_h, K_ * D_ * D_);
  cvt(Wq, Wq_h, D_ * D_);
  cvt(Wk, Wk_h, D_ * D_);
  cvt(Wv, Wv_h, D_ * D_);
  cvt(Wo, Wo_h, D_ * D_);
  basis_transpose<<<(K_ * LP_ + 255) / 256, 256, 0, stream>>>(basis, basisT);
  build_atab<<<(K_ * ND16_ * 32 * 16) / 256, 256, 0, stream>>>(basisT, Atab);
  mgz_kernel<<<(h_ * h_ + 255) / 256, 256, 0, stream>>>(kvs, Wgz, MgzT);

  const int M = B_ * L_;                            // 2048
  const int gx = (D_ / 16) * (M / 16 / 8);          // 32 col-tiles * 16 = 512

  // P_k^T = (x @ Wm_k)^T  (z-batched over k; transposed f16 output)
  wmma_gemm<<<dim3(gx, K_), 256, 0, stream>>>(
      x_h, D_, 0, Wm_h, D_, (size_t)D_ * D_, nullptr, nullptr, P_h, D_, M,
      BLD, nullptr, M, D_, D_);
  // xt = sum_k Toeplitz(basis_k) @ P_k + bm
  combine_kernel<<<(B_ * (L_ / 16) * (D_ / 16)) / 8, 256, 0, stream>>>(
      P_h, Atab, bm, xt_h);
  // q = x@Wq+bq ; k = xt@Wk+bk ; v = xt@Wv+bv (v also transposed)
  wmma_gemm<<<dim3(gx, 1), 256, 0, stream>>>(
      x_h, D_, 0, Wq_h, D_, 0, nullptr, q_h, nullptr, D_, 0, 0, bq, M, D_, D_);
  wmma_gemm<<<dim3(gx, 1), 256, 0, stream>>>(
      xt_h, D_, 0, Wk_h, D_, 0, nullptr, k_h, nullptr, D_, 0, 0, bk, M, D_, D_);
  wmma_gemm<<<dim3(gx, 1), 256, 0, stream>>>(
      xt_h, D_, 0, Wv_h, D_, 0, nullptr, v_h, vT_h, D_, M, 0, bv, M, D_, D_);
  // gates + sim, then cumulative gate sums
  gate_kernel<<<(B_ * H_ * L_) / 8, 256, 0, stream>>>(
      q_h, k_h, v_h, MgzT, bgz, qks, gbuf, simbuf);
  cumsum_kernel<<<B_ * H_, L_, 0, stream>>>(gbuf, cgbuf);
  // causal gated linear attention
  attn_kernel<<<(B_ * H_ * (L_ / 16)) / 8, 256, 0, stream>>>(
      q_h, k_h, vT_h, gbuf, cgbuf, simbuf, kvs, Wg, bg, op_h);
  // out = outpre @ Wo + bo  (fp32 to d_out)
  wmma_gemm<<<dim3(gx, 1), 256, 0, stream>>>(
      op_h, D_, 0, Wo_h, D_, 0, out, nullptr, nullptr, D_, 0, 0, bo, M, D_,
      D_);
}
